// TransformAttentionModel_12833362280586
// MI455X (gfx1250) — compile-verified, matrix operand order checked by reading, not run on hardware
//
#include <hip/hip_runtime.h>

// Problem constants (from reference)
#define BB 8
#define QQ 12
#define PPQ 12
#define NNODES 5000
#define DD 64
#define KHEADS 8
#define DHEAD 8
#define NT 16                                   // nodes per tile
#define NTILES ((NNODES + NT - 1) / NT)         // 313
#define STRIDE 72                               // padded LDS row stride (bf16 elems); 144B -> conflict-free ds_b128, 16B aligned

typedef __attribute__((ext_vector_type(16))) __bf16 v16bf;
typedef __attribute__((ext_vector_type(8)))  __bf16 v8bf;
typedef __attribute__((ext_vector_type(4)))  __bf16 v4bf;
typedef __attribute__((ext_vector_type(2)))  __bf16 v2bf;
typedef __attribute__((ext_vector_type(8)))  float  v8f;
typedef __attribute__((ext_vector_type(4)))  float  v4f;

// ---- f32 -> bf16 helpers ----
// Preference: hw packed cvt > v_perm_b32 pack (round-to-nearest, ties-up) > manual per-element round (ties up).
__device__ __forceinline__ __bf16 f2bf(float f) {
    unsigned u = __builtin_bit_cast(unsigned, f) + 0x8000u;   // round-to-nearest (ties up)
    return __builtin_bit_cast(__bf16, (unsigned short)(u >> 16));
}
__device__ __forceinline__ v2bf f2bf2(float a, float b) {
#if __has_builtin(__builtin_amdgcn_cvt_pk_bf16_f32)
    return __builtin_bit_cast(v2bf, __builtin_amdgcn_cvt_pk_bf16_f32(a, b));
#elif __has_builtin(__builtin_amdgcn_perm)
    unsigned ua = __builtin_bit_cast(unsigned, a) + 0x8000u;
    unsigned ub = __builtin_bit_cast(unsigned, b) + 0x8000u;
    // v_perm_b32: sel codes 0-3 pick src1(=ua) bytes, 4-7 pick src0(=ub) bytes.
    // out = { ua[31:16], ub[31:16] } -> bytes {ua.b2, ua.b3, ub.b2, ub.b3} -> sel 0x07060302
    unsigned r = __builtin_amdgcn_perm(ub, ua, 0x07060302u);
    return __builtin_bit_cast(v2bf, r);
#else
    v2bf y; y.x = f2bf(a); y.y = f2bf(b); return y;
#endif
}
__device__ __forceinline__ v4bf f2bf4(v4f x) {
    v2bf lo = f2bf2(x.x, x.y);
    v2bf hi = f2bf2(x.z, x.w);
    return __builtin_shufflevector(lo, hi, 0, 1, 2, 3);
}
__device__ __forceinline__ float bf2f(__bf16 b) {
    unsigned short h = __builtin_bit_cast(unsigned short, b);
    unsigned u = ((unsigned)h) << 16;
    return __builtin_bit_cast(float, u);
}

// ---- load a 16x32 bf16 A/B fragment from an LDS tile with row stride STRIDE ----
// ISA 16-bit layout: lanes 0-15 hold K {0..7, 16..23}, lanes 16-31 hold K {8..15, 24..31}.
__device__ __forceinline__ v16bf load_frag(const __bf16* base, int lane, int kstep) {
    int m  = lane & 15;
    int c0 = kstep * 32 + ((lane & 16) ? 8 : 0);
    const __bf16* p = base + m * STRIDE + c0;
    v8bf lo = *(const v8bf*)(p);        // K = c0 .. c0+7     (16B aligned)
    v8bf hi = *(const v8bf*)(p + 16);   // K = c0+16 .. c0+23 (16B aligned)
    return __builtin_shufflevector(lo, hi, 0,1,2,3,4,5,6,7,8,9,10,11,12,13,14,15);
}

// ---- stage a [12][nodes<=16][64] f32 slab from global into LDS as bf16 (zero-pad tail nodes) ----
__device__ __forceinline__ void stage_in(const float* __restrict__ g,
                                         const float* __restrict__ gpref,
                                         int n0, int nvalid, __bf16* dst, int tid) {
    for (int v = tid; v < (QQ * NT * DD) / 4; v += 256) {   // 3072 float4 -> 12 iters/thread
        int e = v << 2;
        int r = e >> 10, rem = e & 1023;
        int node = rem >> 6, c = rem & 63;
        v4f x = {0.f, 0.f, 0.f, 0.f};
        if (node < nvalid) {
            int off = (r * NNODES + n0 + node) * DD + c;
            x = *(const v4f*)(g + off);
            if (gpref && ((c & 15) == 0))
                __builtin_prefetch(gpref + off, 0, 1);   // global_prefetch_b8: warm next GEMM's input
        }
        *(v4bf*)(dst + (r * NT + node) * STRIDE + c) = f2bf4(x);
    }
}

// ---- [12*16, 64] x [64, 64]^T GEMM: 48 16x16 output tiles over 8 waves, WMMA bf16 ----
// nt = wave&3 is per-wave constant -> compiler hoists weight (B) fragments out of the tile loop.
__device__ __forceinline__ void gemm_to_lds(const __bf16* __restrict__ in,
                                            const __bf16* __restrict__ w,
                                            const float*  __restrict__ bias,
                                            __bf16* __restrict__ out,
                                            int wave, int lane, bool relu) {
#pragma unroll 2
    for (int t = wave; t < 48; t += 8) {
        int rg = t >> 2, nt = t & 3;
        v8f acc = {};
        const __bf16* abase = in + rg * 16 * STRIDE;
        const __bf16* bbase = w  + nt * 16 * STRIDE;
#pragma unroll
        for (int ks = 0; ks < 2; ++ks) {
            v16bf a = load_frag(abase, lane, ks);
            v16bf b = load_frag(bbase, lane, ks);
            acc = __builtin_amdgcn_wmma_f32_16x16x32_bf16(false, a, false, b,
                                                          (short)0, acc, false, false);
        }
        int col = nt * 16 + (lane & 15);
        float bv = bias[col];
        int rofs = (lane & 16) ? 8 : 0;
#pragma unroll
        for (int i = 0; i < 8; i += 2) {
            float v0 = acc[i] + bv;
            float v1 = acc[i + 1] + bv;
            if (relu) { v0 = fmaxf(v0, 0.f); v1 = fmaxf(v1, 0.f); }
            v2bf p = f2bf2(v0, v1);
            out[(rg * 16 + i + rofs)     * STRIDE + col] = p.x;
            out[(rg * 16 + i + 1 + rofs) * STRIDE + col] = p.y;
        }
    }
}

__global__ __launch_bounds__(256)
void fused_sta_kernel(const float* __restrict__ X,   const float* __restrict__ SP,
                      const float* __restrict__ SQ,
                      const float* __restrict__ W21, const float* __restrict__ b21,
                      const float* __restrict__ W22, const float* __restrict__ b22,
                      const float* __restrict__ W23, const float* __restrict__ b23,
                      const float* __restrict__ W24, const float* __restrict__ b24,
                      const float* __restrict__ W25, const float* __restrict__ b25,
                      float* __restrict__ out) {
    extern __shared__ char smem[];
    // LDS carve (bytes): W 46080 | bias 1280 | stage/O 27648 | Qs/H 27648 | Ks 27648 | Vs 27648
    __bf16* Wl  = (__bf16*)(smem);
    float*  bl  = (float*)(smem + 46080);
    __bf16* stg = (__bf16*)(smem + 47360);               // staged input, later attention output O
    __bf16* qs  = (__bf16*)(smem + 47360 + 27648);       // Q-hat, later FFN hidden H
    __bf16* ks  = (__bf16*)(smem + 47360 + 2 * 27648);   // K-hat
    __bf16* vs  = (__bf16*)(smem + 47360 + 3 * 27648);   // V-hat

    const int tid  = threadIdx.x;
    const int lane = tid & 31;
    const int wave = tid >> 5;

    const int blk = blockIdx.x;
    const int b   = blk / NTILES;
    const int t   = blk - b * NTILES;
    const int n0  = t * NT;
    const int nv  = (NNODES - n0) < NT ? (NNODES - n0) : NT;

    // ---- Phase 0: weights + biases -> LDS (bf16 / f32), float4 reads + packed cvt ----
    const float* Ws[5] = {W21, W22, W23, W24, W25};
    const float* Bs[5] = {b21, b22, b23, b24, b25};
#pragma unroll
    for (int w = 0; w < 5; ++w)
        for (int v = tid; v < 1024; v += 256) {          // 4096 f32 = 1024 float4
            int idx = v << 2;
            int j = idx >> 6, c = idx & 63;
            v4f x = *(const v4f*)(Ws[w] + idx);
            *(v4bf*)(Wl + (w * 64 + j) * STRIDE + c) = f2bf4(x);
        }
    for (int idx = tid; idx < 5 * 64; idx += 256)
        bl[idx] = Bs[idx >> 6][idx & 63];
    __syncthreads();

    const float* SQb = SQ + (size_t)b * QQ  * NNODES * DD;
    const float* SPb = SP + (size_t)b * PPQ * NNODES * DD;
    const float* Xb  = X  + (size_t)b * PPQ * NNODES * DD;

    // ---- Phase 1: Q-hat = relu(SQ @ W21^T + b21) ----
    stage_in(SQb, SPb, n0, nv, stg, tid);
    __syncthreads();
    gemm_to_lds(stg, Wl + 0 * 64 * STRIDE, bl + 0 * 64, qs, wave, lane, true);
    __syncthreads();

    // ---- Phase 2: K-hat = relu(SP @ W22^T + b22) ----
    stage_in(SPb, Xb, n0, nv, stg, tid);
    __syncthreads();
    gemm_to_lds(stg, Wl + 1 * 64 * STRIDE, bl + 1 * 64, ks, wave, lane, true);
    __syncthreads();

    // ---- Phase 3: V-hat = relu(X @ W23^T + b23) ----
    stage_in(Xb, (const float*)0, n0, nv, stg, tid);
    __syncthreads();
    gemm_to_lds(stg, Wl + 2 * 64 * STRIDE, bl + 2 * 64, vs, wave, lane, true);
    __syncthreads();

    // ---- Phase 4: attention, thread = (node, head, q-half); softmax rows are thread-local ----
    {
        const int nd = tid & 15;
        const int h  = (tid >> 4) & 7;
        const int u  = tid >> 7;
        const float scale = 0.35355339059327373f;   // 1/sqrt(8)
        for (int qi = u * 6; qi < u * 6 + 6; ++qi) {
            v8bf qv8 = *(const v8bf*)(qs + (qi * NT + nd) * STRIDE + h * DHEAD);
            float qv[8];
#pragma unroll
            for (int d = 0; d < 8; ++d) qv[d] = bf2f(qv8[d]);
            float sc[12], m = -1e30f;
#pragma unroll
            for (int pi = 0; pi < 12; ++pi) {
                v8bf kv8 = *(const v8bf*)(ks + (pi * NT + nd) * STRIDE + h * DHEAD);
                float s = 0.f;
#pragma unroll
                for (int d = 0; d < 8; ++d) s += qv[d] * bf2f(kv8[d]);
                s *= scale;
                sc[pi] = s;
                m = fmaxf(m, s);
            }
            float sum = 0.f;
#pragma unroll
            for (int pi = 0; pi < 12; ++pi) { sc[pi] = __expf(sc[pi] - m); sum += sc[pi]; }
            float inv = 1.0f / sum;
            float acc[8] = {0.f, 0.f, 0.f, 0.f, 0.f, 0.f, 0.f, 0.f};
#pragma unroll
            for (int pi = 0; pi < 12; ++pi) {
                v8bf vv8 = *(const v8bf*)(vs + (pi * NT + nd) * STRIDE + h * DHEAD);
                float w = sc[pi] * inv;
#pragma unroll
                for (int d = 0; d < 8; ++d) acc[d] += w * bf2f(vv8[d]);
            }
            v4f lo = {acc[0], acc[1], acc[2], acc[3]};
            v4f hi = {acc[4], acc[5], acc[6], acc[7]};
            v8bf ov = __builtin_shufflevector(f2bf4(lo), f2bf4(hi), 0, 1, 2, 3, 4, 5, 6, 7);
            *(v8bf*)(stg + (qi * NT + nd) * STRIDE + h * DHEAD) = ov;   // O tile reuses stage buffer
        }
    }
    __syncthreads();

    // ---- Phase 5: H = relu(O @ W24^T + b24) (H reuses Q-hat buffer) ----
    gemm_to_lds(stg, Wl + 3 * 64 * STRIDE, bl + 3 * 64, qs, wave, lane, true);
    __syncthreads();

    // ---- Phase 6: out = H @ W25^T + b25 -> global f32 ----
#pragma unroll 2
    for (int tt = wave; tt < 48; tt += 8) {
        int rg = tt >> 2, nt = tt & 3;
        v8f acc = {};
        const __bf16* abase = qs + rg * 16 * STRIDE;
        const __bf16* bbase = Wl + 4 * 64 * STRIDE + nt * 16 * STRIDE;
#pragma unroll
        for (int ksi = 0; ksi < 2; ++ksi) {
            v16bf a = load_frag(abase, lane, ksi);
            v16bf bm = load_frag(bbase, lane, ksi);
            acc = __builtin_amdgcn_wmma_f32_16x16x32_bf16(false, a, false, bm,
                                                          (short)0, acc, false, false);
        }
        int col = nt * 16 + (lane & 15);
        float bv = bl[4 * 64 + col];
        int rofs = (lane & 16) ? 8 : 0;
#pragma unroll
        for (int i = 0; i < 8; ++i) {
            int node = i + rofs;
            if (node < nv)
                out[(((size_t)b * QQ + rg) * NNODES + n0 + node) * DD + col] = acc[i] + bv;
        }
    }
}

extern "C" void kernel_launch(void* const* d_in, const int* in_sizes, int n_in,
                              void* d_out, int out_size, void* d_ws, size_t ws_size,
                              hipStream_t stream) {
    (void)in_sizes; (void)n_in; (void)out_size; (void)d_ws; (void)ws_size;
    const float* X   = (const float*)d_in[0];
    const float* SP  = (const float*)d_in[1];
    const float* SQ  = (const float*)d_in[2];
    const float* W21 = (const float*)d_in[3];
    const float* b21 = (const float*)d_in[4];
    const float* W22 = (const float*)d_in[5];
    const float* b22 = (const float*)d_in[6];
    const float* W23 = (const float*)d_in[7];
    const float* b23 = (const float*)d_in[8];
    const float* W24 = (const float*)d_in[9];
    const float* b24 = (const float*)d_in[10];
    const float* W25 = (const float*)d_in[11];
    const float* b25 = (const float*)d_in[12];
    float* out = (float*)d_out;

    const size_t lds_bytes = 46080 + 1280 + 4 * 27648;   // 157,952 B -> 2 WGs per 320KB WGP
    hipFuncSetAttribute((const void*)fused_sta_kernel,
                        hipFuncAttributeMaxDynamicSharedMemorySize, (int)lds_bytes);

    dim3 grid(BB * NTILES);   // 8 * 313 = 2504 workgroups
    fused_sta_kernel<<<grid, 256, lds_bytes, stream>>>(X, SP, SQ,
                                                       W21, b21, W22, b22, W23, b23,
                                                       W24, b24, W25, b25, out);
}
